// PointVoxelCNN_12232066859479
// MI455X (gfx1250) — matrix operand order repaired and checked by reading, not compiled, not run on hardware
//
#include <hip/hip_runtime.h>
#include <hip/hip_bf16.h>

// ---------------------------------------------------------------------------
// PointVoxelCNN block for MI455X (gfx1250, wave32, WMMA)
// conv3d / conv1x1 as implicit GEMM on v_wmma_f32_16x16x32_bf16.
// Per tap: batch-issue 2 A-fragments + 8 B-fragments, then 8 WMMAs on
// 4 independent accumulator chains -> partial loadcnt waits, load/WMMA overlap.
// ---------------------------------------------------------------------------

#define RES 32
#define VOX (RES * RES * RES)   // 32768
#define NPT 32768
#define NB  8
#define CH  64
#define GRP 32
#define INV_SQRT2 0.70710678118654752f

typedef __attribute__((ext_vector_type(16))) __bf16 v16bf;
typedef __attribute__((ext_vector_type(8)))  float  v8f;

union Frag16 { v16bf v; uint4 q[2]; };

__device__ __forceinline__ unsigned short f2bf(float f) {
  union { float f; unsigned u; } x; x.f = f;
  unsigned u = x.u;
  unsigned r = (u + 0x7FFFu + ((u >> 16) & 1u)) >> 16;  // round-to-nearest-even
  return (unsigned short)r;
}
__device__ __forceinline__ float lrelu(float x) { return x > 0.f ? x : 0.01f * x; }

// ---------------------------------------------------------------------------
// zero fill
// ---------------------------------------------------------------------------
__global__ __launch_bounds__(256) void k_zero(float* p, size_t n) {
  size_t i = (size_t)blockIdx.x * blockDim.x + threadIdx.x;
  size_t stride = (size_t)gridDim.x * blockDim.x;
  for (; i < n; i += stride) p[i] = 0.f;
}

// ---------------------------------------------------------------------------
// weight packing into WMMA B-fragment order (bf16)
// conv3d weights [O=64][I=64][3][3][3] -> packB [tap=27][kc=2][nt=4][lane=32][e=16]
// B-fragment (16x16x32 bf16): lane L -> column n = L%16 (+16*nt),
// element e -> K = (e&7) + (e>>3)*16 + (L>=16 ? 8 : 0); in-channel = kc*32 + K.
// ---------------------------------------------------------------------------
__global__ __launch_bounds__(256) void k_pack_w3(const float* __restrict__ w,
                                                 unsigned short* __restrict__ packB) {
  const int total = 27 * 2 * 4 * 32 * 16;
  for (int i = blockIdx.x * blockDim.x + threadIdx.x; i < total;
       i += gridDim.x * blockDim.x) {
    int e    = i & 15;
    int lane = (i >> 4) & 31;
    int nt   = (i >> 9) & 3;
    int kc   = (i >> 11) & 1;
    int tap  = i >> 12;
    int half = lane >> 4;
    int n    = nt * 16 + (lane & 15);
    int k    = kc * 32 + (e & 7) + ((e >> 3) << 4) + half * 8;
    packB[i] = f2bf(w[((size_t)n * 64 + k) * 27 + tap]);
  }
}

// 1x1 weights [O=64][I=64] -> packB [kc=2][nt=4][lane=32][e=16]
__global__ __launch_bounds__(256) void k_pack_w1(const float* __restrict__ w,
                                                 unsigned short* __restrict__ packB) {
  const int total = 2 * 4 * 32 * 16;
  for (int i = blockIdx.x * blockDim.x + threadIdx.x; i < total;
       i += gridDim.x * blockDim.x) {
    int e    = i & 15;
    int lane = (i >> 4) & 31;
    int nt   = (i >> 9) & 3;
    int kc   = i >> 11;
    int half = lane >> 4;
    int n    = nt * 16 + (lane & 15);
    int k    = kc * 32 + (e & 7) + ((e >> 3) << 4) + half * 8;
    packB[i] = f2bf(w[(size_t)n * 64 + k]);
  }
}

// ---------------------------------------------------------------------------
// trilinear scatter: 64 threads per point (one channel each).
// block = 256 threads = 4 points.
// ---------------------------------------------------------------------------
__global__ __launch_bounds__(256) void k_voxelize(const float* __restrict__ points,
                                                  const float* __restrict__ feats,
                                                  float* __restrict__ featsum,
                                                  float* __restrict__ wsum) {
  int pid = blockIdx.x * 4 + (threadIdx.x >> 6);     // global point id
  int c   = threadIdx.x & 63;                        // channel
  int b = pid >> 15, n = pid & (NPT - 1);
  const float* pp = points + ((size_t)b * NPT + n) * 3;
  float cx = fminf(fmaxf(pp[0], 0.f), 1.f) * (RES - 1);
  float cy = fminf(fmaxf(pp[1], 0.f), 1.f) * (RES - 1);
  float cz = fminf(fmaxf(pp[2], 0.f), 1.f) * (RES - 1);
  int ix = (int)floorf(cx), iy = (int)floorf(cy), iz = (int)floorf(cz);
  float fx = cx - ix, fy = cy - iy, fz = cz - iz;
  float fv = feats[((size_t)b * NPT + n) * CH + c];
  #pragma unroll
  for (int j = 0; j < 8; ++j) {
    int dx = (j >> 2) & 1, dy = (j >> 1) & 1, dz = j & 1;
    int gx = min(ix + dx, RES - 1), gy = min(iy + dy, RES - 1), gz = min(iz + dz, RES - 1);
    float w = (dx ? fx : 1.f - fx) * (dy ? fy : 1.f - fy) * (dz ? fz : 1.f - fz);
    size_t lin = ((size_t)gx * RES + gy) * RES + gz;
    if (c == 0) atomicAdd(&wsum[(size_t)b * VOX + lin], w);
    atomicAdd(&featsum[((size_t)b * VOX + lin) * CH + c], w * fv);
  }
}

// ---------------------------------------------------------------------------
// GroupNorm (32 groups of 2ch) over channels-last [B][V][64].
// If wsum != nullptr, src is featsum and is divided by max(wsum,1e-8).
// Writes optional f32 normalized tensor and bf16 leaky_relu activations.
// ---------------------------------------------------------------------------
__global__ __launch_bounds__(256) void k_gnorm(const float* __restrict__ src,
                                               const float* __restrict__ wsum,
                                               const float* __restrict__ gamma,
                                               const float* __restrict__ beta,
                                               float* __restrict__ norm_out,
                                               unsigned short* __restrict__ act_out,
                                               int V) {
  int b = blockIdx.x / GRP, g = blockIdx.x % GRP;
  int tid = threadIdx.x;
  const float* sb = src + (size_t)b * V * CH;
  const float* wb = wsum ? wsum + (size_t)b * V : nullptr;

  float s1 = 0.f, s2 = 0.f;
  for (int v = tid; v < V; v += 256) {
    const float2 xy = *(const float2*)(sb + (size_t)v * CH + 2 * g);
    float a = xy.x, c = xy.y;
    if (wb) { float iw = 1.f / fmaxf(wb[v], 1e-8f); a *= iw; c *= iw; }
    s1 += a + c; s2 += a * a + c * c;
  }
  __shared__ float r1[256], r2[256];
  r1[tid] = s1; r2[tid] = s2; __syncthreads();
  for (int off = 128; off > 0; off >>= 1) {
    if (tid < off) { r1[tid] += r1[tid + off]; r2[tid] += r2[tid + off]; }
    __syncthreads();
  }
  __shared__ float mu_s, rs_s;
  if (tid == 0) {
    float cnt = 2.f * (float)V;
    float mu = r1[0] / cnt;
    float var = r2[0] / cnt - mu * mu;
    mu_s = mu; rs_s = rsqrtf(var + 1e-5f);
  }
  __syncthreads();
  float mu = mu_s, rs = rs_s;
  float g0 = gamma[2 * g], g1 = gamma[2 * g + 1];
  float b0 = beta[2 * g],  b1 = beta[2 * g + 1];

  for (int v = tid; v < V; v += 256) {
    size_t base = (size_t)v * CH + 2 * g;
    const float2 xy = *(const float2*)(sb + base);
    float a = xy.x, c = xy.y;
    if (wb) { float iw = 1.f / fmaxf(wb[v], 1e-8f); a *= iw; c *= iw; }
    float y0 = (a - mu) * rs * g0 + b0;
    float y1 = (c - mu) * rs * g1 + b1;
    size_t o = (size_t)b * V * CH + base;
    if (norm_out) { norm_out[o] = y0; norm_out[o + 1] = y1; }
    act_out[o]     = f2bf(lrelu(y0));
    act_out[o + 1] = f2bf(lrelu(y1));
  }
}

// ---------------------------------------------------------------------------
// conv3d 3x3x3 SAME, 64->64 ch, WMMA implicit GEMM.
// Block (256 thr = 8 waves) covers (b, x, 4 consecutive y).
// Wave -> (y = y4*4 + wv>>1, z-subtile msub = wv&1), computes
// 16 voxels x 64 out-ch with 4 accumulator chains (4x A reuse).
// OOB taps read from a zeroed pad row (branchless address select).
// Per tap: all 10 fragment loads issued before the 8 WMMAs.
// fuse_residual: out[idx] = (acc + bias + out[idx]) * 1/sqrt(2)  (in place)
// ---------------------------------------------------------------------------
__global__ __launch_bounds__(256) void k_conv3_wmma(const unsigned short* __restrict__ act,
                                                    const unsigned short* __restrict__ packB,
                                                    const float* __restrict__ bias,
                                                    const unsigned short* __restrict__ zeropad,
                                                    float* __restrict__ out,
                                                    int fuse_residual) {
  int b   = blockIdx.x >> 8;
  int rem = blockIdx.x & 255;
  int x   = rem >> 3;
  int y4  = rem & 7;
  int tid = threadIdx.x;
  int wv = tid >> 5, lane = tid & 31;
  int half = lane >> 4, l16 = lane & 15;
  int y  = y4 * 4 + (wv >> 1);
  int z0 = (wv & 1) * 16;
  int m  = l16;                        // A-fragment row of this lane

  const unsigned short* actb = act + (size_t)b * VOX * CH;
  v8f acc[4];
  #pragma unroll
  for (int nt = 0; nt < 4; ++nt)
    acc[nt] = (v8f){0.f, 0.f, 0.f, 0.f, 0.f, 0.f, 0.f, 0.f};

  for (int kd = 0; kd < 3; ++kd) {
    int xx = x + kd - 1;
    bool xok = (unsigned)xx < (unsigned)RES;
    // prefetch next-kd center row of the activation grid
    {
      int xn = min(max(x + kd, 0), RES - 1);
      const unsigned short* pf =
          actb + (((size_t)xn * RES + y) * RES + (z0 + m)) * CH;
      __builtin_prefetch(pf, 0, 3);
    }
    #pragma unroll
    for (int kh = 0; kh < 3; ++kh) {
      int yy = y + kh - 1;
      bool yok = (unsigned)yy < (unsigned)RES;
      #pragma unroll
      for (int kw = 0; kw < 3; ++kw) {
        int zz = z0 + m + kw - 1;
        bool ok = xok && yok && ((unsigned)zz < (unsigned)RES);
        const unsigned short* ap =
            ok ? actb + (((size_t)xx * RES + yy) * RES + zz) * CH : zeropad;
        int tap = (kd * 3 + kh) * 3 + kw;
        const unsigned short* bbase =
            packB + ((size_t)tap * 2) * 4 * 32 * 16;

        // ---- batch-issue all loads for this tap ----
        Frag16 afr[2];
        #pragma unroll
        for (int kc = 0; kc < 2; ++kc) {
          const uint4* pa = (const uint4*)(ap + kc * 32 + half * 8);
          afr[kc].q[0] = pa[0];        // in-ch base .. base+7
          afr[kc].q[1] = pa[2];        // in-ch base+16 .. base+23
        }
        Frag16 bfr[8];
        #pragma unroll
        for (int kc = 0; kc < 2; ++kc) {
          #pragma unroll
          for (int nt = 0; nt < 4; ++nt) {
            const uint4* pb = (const uint4*)(bbase +
                (((size_t)kc * 4 + nt) * 32 + lane) * 16);
            bfr[kc * 4 + nt].q[0] = pb[0];
            bfr[kc * 4 + nt].q[1] = pb[1];
          }
        }
        // ---- 8 WMMAs, 4 independent accumulator chains ----
        #pragma unroll
        for (int kc = 0; kc < 2; ++kc) {
          #pragma unroll
          for (int nt = 0; nt < 4; ++nt) {
            acc[nt] = __builtin_amdgcn_wmma_f32_16x16x32_bf16(
                false, afr[kc].v, false, bfr[kc * 4 + nt].v,
                (short)0, acc[nt], false, false);
          }
        }
      }
    }
  }

  float* ob = out + (size_t)b * VOX * CH;
  #pragma unroll
  for (int nt = 0; nt < 4; ++nt) {
    int n = nt * 16 + l16;
    float bn = bias[n];
    #pragma unroll
    for (int r = 0; r < 8; ++r) {
      int mz = r + half * 8;           // D row -> voxel z = z0 + mz
      size_t idx = (((size_t)x * RES + y) * RES + (z0 + mz)) * CH + n;
      float val = acc[nt][r] + bn;
      if (fuse_residual) val = (val + ob[idx]) * INV_SQRT2;
      ob[idx] = val;
    }
  }
}

// ---------------------------------------------------------------------------
// 1x1 conv (64->64) over points via WMMA; wave = 16 points x 64 out-ch.
// block 256 thr = 8 waves = 128 points.  All loads issued before the WMMAs.
// mode 0: out = acc + bias
// mode 1: out[idx] = (acc+bias+fnorm[idx])*0.5 + out[idx]*1/sqrt(2)
// ---------------------------------------------------------------------------
__global__ __launch_bounds__(256) void k_pconv_wmma(const unsigned short* __restrict__ act,
                                                    const unsigned short* __restrict__ packB,
                                                    const float* __restrict__ bias,
                                                    const float* __restrict__ fnorm,
                                                    float* __restrict__ out,
                                                    int mode) {
  int b     = blockIdx.x >> 8;
  int ptile = blockIdx.x & 255;        // 128 points per block
  int tid = threadIdx.x;
  int wv = tid >> 5, lane = tid & 31;
  int half = lane >> 4, l16 = lane & 15;
  int p0 = ptile * 128 + wv * 16;

  const unsigned short* ap = act + ((size_t)b * NPT + p0 + l16) * CH;
  v8f acc[4];
  #pragma unroll
  for (int nt = 0; nt < 4; ++nt)
    acc[nt] = (v8f){0.f, 0.f, 0.f, 0.f, 0.f, 0.f, 0.f, 0.f};

  // ---- batch-issue all loads ----
  Frag16 afr[2];
  #pragma unroll
  for (int kc = 0; kc < 2; ++kc) {
    const uint4* pa = (const uint4*)(ap + kc * 32 + half * 8);
    afr[kc].q[0] = pa[0];
    afr[kc].q[1] = pa[2];
  }
  Frag16 bfr[8];
  #pragma unroll
  for (int kc = 0; kc < 2; ++kc) {
    #pragma unroll
    for (int nt = 0; nt < 4; ++nt) {
      const uint4* pb = (const uint4*)(packB +
          (((size_t)kc * 4 + nt) * 32 + lane) * 16);
      bfr[kc * 4 + nt].q[0] = pb[0];
      bfr[kc * 4 + nt].q[1] = pb[1];
    }
  }
  // ---- 8 WMMAs ----
  #pragma unroll
  for (int kc = 0; kc < 2; ++kc) {
    #pragma unroll
    for (int nt = 0; nt < 4; ++nt) {
      acc[nt] = __builtin_amdgcn_wmma_f32_16x16x32_bf16(
          false, afr[kc].v, false, bfr[kc * 4 + nt].v,
          (short)0, acc[nt], false, false);
    }
  }

  #pragma unroll
  for (int nt = 0; nt < 4; ++nt) {
    int n = nt * 16 + l16;
    float bn = bias[n];
    #pragma unroll
    for (int r = 0; r < 8; ++r) {
      int mp = r + half * 8;
      size_t idx = ((size_t)b * NPT + p0 + mp) * CH + n;
      float val = acc[nt][r] + bn;
      if (mode == 1) val = (val + fnorm[idx]) * 0.5f + out[idx] * INV_SQRT2;
      out[idx] = val;
    }
  }
}

// ---------------------------------------------------------------------------
// trilinear gather from out_grid (f32 [B][VOX][64]) -> d_out
// ---------------------------------------------------------------------------
__global__ __launch_bounds__(256) void k_devox(const float* __restrict__ points,
                                               const float* __restrict__ grid,
                                               float* __restrict__ out) {
  int idx = blockIdx.x * blockDim.x + threadIdx.x;
  if (idx >= NB * NPT) return;
  int b = idx >> 15, n = idx & (NPT - 1);
  const float* pp = points + ((size_t)b * NPT + n) * 3;
  float cx = fminf(fmaxf(pp[0], 0.f), 1.f) * (RES - 1);
  float cy = fminf(fmaxf(pp[1], 0.f), 1.f) * (RES - 1);
  float cz = fminf(fmaxf(pp[2], 0.f), 1.f) * (RES - 1);
  int ix = (int)floorf(cx), iy = (int)floorf(cy), iz = (int)floorf(cz);
  float fx = cx - ix, fy = cy - iy, fz = cz - iz;

  float4 acc[16];
  #pragma unroll
  for (int t = 0; t < 16; ++t) acc[t] = make_float4(0.f, 0.f, 0.f, 0.f);

  for (int j = 0; j < 8; ++j) {
    int dx = (j >> 2) & 1, dy = (j >> 1) & 1, dz = j & 1;
    int gx = min(ix + dx, RES - 1), gy = min(iy + dy, RES - 1), gz = min(iz + dz, RES - 1);
    float w = (dx ? fx : 1.f - fx) * (dy ? fy : 1.f - fy) * (dz ? fz : 1.f - fz);
    size_t lin = ((size_t)gx * RES + gy) * RES + gz;
    const float4* gp = (const float4*)(grid + ((size_t)b * VOX + lin) * CH);
    #pragma unroll
    for (int t = 0; t < 16; ++t) {
      float4 v = gp[t];
      acc[t].x += w * v.x; acc[t].y += w * v.y;
      acc[t].z += w * v.z; acc[t].w += w * v.w;
    }
  }
  float4* op = (float4*)(out + ((size_t)b * NPT + n) * CH);
  #pragma unroll
  for (int t = 0; t < 16; ++t) op[t] = acc[t];
}

// ---------------------------------------------------------------------------
// launch
// ---------------------------------------------------------------------------
extern "C" void kernel_launch(void* const* d_in, const int* in_sizes, int n_in,
                              void* d_out, int out_size, void* d_ws, size_t ws_size,
                              hipStream_t stream) {
  const float* points   = (const float*)d_in[0];
  const float* features = (const float*)d_in[1];
  const float* conv1_w  = (const float*)d_in[2];
  const float* conv1_b  = (const float*)d_in[3];
  const float* conv2_w  = (const float*)d_in[4];
  const float* conv2_b  = (const float*)d_in[5];
  const float* g1_gamma = (const float*)d_in[6];
  const float* g1_beta  = (const float*)d_in[7];
  const float* g2_gamma = (const float*)d_in[8];
  const float* g2_beta  = (const float*)d_in[9];
  const float* pc1_w    = (const float*)d_in[10];
  const float* pc1_b    = (const float*)d_in[11];
  const float* pc2_w    = (const float*)d_in[12];
  const float* pc2_b    = (const float*)d_in[13];
  const float* pg1_gamma= (const float*)d_in[14];
  const float* pg1_beta = (const float*)d_in[15];
  const float* pg2_gamma= (const float*)d_in[16];
  const float* pg2_beta = (const float*)d_in[17];

  const size_t szGrid = (size_t)NB * VOX * CH * sizeof(float);   // 64 MB
  const size_t szW    = (size_t)NB * VOX * sizeof(float);        //  1 MB
  const size_t szPad  = 64 * sizeof(float);                      // 256 B zero pad
  const size_t szAct  = (size_t)NB * VOX * CH * sizeof(unsigned short); // 32 MB

  char* w = (char*)d_ws;
  size_t o = 0;
  float*          bufA     = (float*)(w + o);          o += szGrid;  // featsum / conv1out / h1
  float*          wsum     = (float*)(w + o);          o += szW;
  unsigned short* zeropad  = (unsigned short*)(w + o); o += szPad;   // zeroed with featsum+wsum
  float*          gridnorm = (float*)(w + o);          o += szGrid;  // -> out_grid in place
  float*          fnorm    = (float*)(w + o);          o += szGrid;
  unsigned short* actV     = (unsigned short*)(w + o); o += szAct;   // act1 / act2 (voxel)
  unsigned short* actP     = (unsigned short*)(w + o); o += szAct;   // pact1 / pact2
  unsigned short* packB1   = (unsigned short*)(w + o); o += 27*2*4*32*16*2;
  unsigned short* packB2   = (unsigned short*)(w + o); o += 27*2*4*32*16*2;
  unsigned short* packBp1  = (unsigned short*)(w + o); o += 2*4*32*16*2;
  unsigned short* packBp2  = (unsigned short*)(w + o); o += 2*4*32*16*2;
  (void)ws_size; (void)n_in; (void)in_sizes; (void)out_size;

  float* outp = (float*)d_out;

  // 0) zero featsum + wsum + pad (contiguous)
  size_t nzero = (size_t)NB * VOX * CH + (size_t)NB * VOX + 64;
  k_zero<<<4096, 256, 0, stream>>>(bufA, nzero);

  // weight packing (tiny)
  k_pack_w3<<<432, 256, 0, stream>>>(conv1_w, packB1);
  k_pack_w3<<<432, 256, 0, stream>>>(conv2_w, packB2);
  k_pack_w1<<<16, 256, 0, stream>>>(pc1_w, packBp1);
  k_pack_w1<<<16, 256, 0, stream>>>(pc2_w, packBp2);

  // 1) voxelize (scatter, 64 threads per point)
  k_voxelize<<<(NB * NPT) / 4, 256, 0, stream>>>(points, features, bufA, wsum);

  // 2) GN1 (divide by wsum fused) -> gridnorm f32 + actV bf16 (lrelu)
  k_gnorm<<<NB * GRP, 256, 0, stream>>>(bufA, wsum, g1_gamma, g1_beta,
                                        gridnorm, actV, VOX);

  // 3) conv1 (WMMA) -> bufA
  k_conv3_wmma<<<NB * 32 * 8, 256, 0, stream>>>(actV, packB1, conv1_b, zeropad,
                                                bufA, 0);

  // 4) GN2 -> actV bf16
  k_gnorm<<<NB * GRP, 256, 0, stream>>>(bufA, nullptr, g2_gamma, g2_beta,
                                        nullptr, actV, VOX);

  // 5) conv2 (WMMA) + residual: gridnorm <- (conv2+gridnorm)/sqrt2 (= out_grid)
  k_conv3_wmma<<<NB * 32 * 8, 256, 0, stream>>>(actV, packB2, conv2_b, zeropad,
                                                gridnorm, 1);

  // 6) point GN1 -> fnorm f32 + actP bf16
  k_gnorm<<<NB * GRP, 256, 0, stream>>>(features, nullptr, pg1_gamma, pg1_beta,
                                        fnorm, actP, NPT);

  // 7) point conv1 (WMMA) -> bufA (h1)
  k_pconv_wmma<<<NB * (NPT / 128), 256, 0, stream>>>(actP, packBp1, pc1_b,
                                                     nullptr, bufA, 0);

  // 8) point GN2 -> actP bf16
  k_gnorm<<<NB * GRP, 256, 0, stream>>>(bufA, nullptr, pg2_gamma, pg2_beta,
                                        nullptr, actP, NPT);

  // 9) devoxelize out_grid -> d_out (staging devox term)
  k_devox<<<(NB * NPT) / 256, 256, 0, stream>>>(points, gridnorm, outp);

  // 10) point conv2 (WMMA) final fuse: d_out = (h2 + fnorm)/2 + devox/sqrt2
  k_pconv_wmma<<<NB * (NPT / 128), 256, 0, stream>>>(actP, packBp2, pc2_b,
                                                     fnorm, outp, 1);
}
